// RLIFLayer_37074157699153
// MI455X (gfx1250) — matrix operand order repaired and checked by reading, not compiled, not run on hardware
//
#include <hip/hip_runtime.h>
#include <hip/hip_bf16.h>

// RLIF layer for MI455X (gfx1250, wave32, WMMA).
// T=16, B=1024, H=1024, THRESH=DECAY=0.5.
// One WG = 16 batch rows x all 1024 hidden outputs, iterated over all 16
// timesteps with vmem resident in VGPRs (WMMA C layout). Spikes are binary ->
// exact in bf16; W is pre-packed to bf16 scaled by THRESH so the recurrent
// matmul runs on v_wmma_f32_16x16x32_bf16. W (2MB bf16) stays in the 192MB L2.
// 512 threads = 16 waves, 4 column tiles per wave: low enough VGPR pressure
// to double-buffer B tiles in registers (loads issued one kt ahead of use)
// WITHOUT spilling, plus 4 waves/SIMD of occupancy for latency hiding.

typedef __bf16 bf16_t;
typedef __bf16 v8bf  __attribute__((ext_vector_type(8)));   // 16B = 4 VGPRs
typedef __bf16 v16bf __attribute__((ext_vector_type(16)));  // 32B = 8 VGPRs
typedef float  v8f   __attribute__((ext_vector_type(8)));   // 8 VGPRs

#define T_STEPS 16
#define BATCH   1024
#define HID     1024
#define THRESH  0.5f
#define DECAY   0.5f
#define THREADS 512
#define NTILES  4   // 16 waves * 4 tiles * 16 cols = 1024 outputs

// ---- pack kernel: Wt[n][k] = bf16(THRESH * W[n][k]) --------------------------
__global__ void rlif_pack_w(const float* __restrict__ W,
                            bf16_t* __restrict__ Wt, int n) {
    int i = blockIdx.x * blockDim.x + threadIdx.x;
    if (i < n) Wt[i] = (bf16_t)(THRESH * W[i]);
}

// ---- main recurrent kernel ---------------------------------------------------
__global__ __launch_bounds__(THREADS)
void rlif_step_kernel(const float* __restrict__ x,      // [T,B,H] f32
                      const bf16_t* __restrict__ Wt,    // [H,H] bf16, = 0.5*W
                      const float* __restrict__ bias,   // [H] f32
                      float* __restrict__ out) {        // [T,B,H] f32
    // Previous-step binary spikes for this WG's 16 batch rows: 16 x 1024 bf16.
    __shared__ bf16_t spikes[16 * HID];                 // 32 KB of 320 KB LDS

    const int tid   = threadIdx.x;
    const int wave  = tid >> 5;        // 0..15 (wave32)
    const int lane  = tid & 31;
    const int lhalf = lane >> 4;       // 0: lanes 0-15, 1: lanes 16-31
    const int lmod  = lane & 15;
    const int bbase = blockIdx.x * 16; // batch-row block

    // Per-lane base into packed W for this wave's column tiles:
    // B layout (32x16 bf16): lanes 0-15 hold K 0..15 of col N=lmod,
    // lanes 16-31 hold K 16..31 -> 32 contiguous bytes per lane per tile.
    const bf16_t* wbase =
        Wt + (size_t)(wave * (NTILES * 16) + lmod) * HID + lhalf * 16;

    // t=0 has no prior spikes: zero the LDS (512 threads x 64B).
    {
        uint4* p = (uint4*)spikes;
        #pragma unroll
        for (int i = 0; i < 4; ++i)
            p[tid + i * THREADS] = make_uint4(0u, 0u, 0u, 0u);
    }
    __syncthreads();

    // Persistent vmem in WMMA C layout: VGPR r holds (M = r + 8*lhalf, N = lmod).
    v8f acc[NTILES] = {};              // vmem starts at 0
    float bb[NTILES];
    #pragma unroll
    for (int ti = 0; ti < NTILES; ++ti)
        bb[ti] = bias[wave * (NTILES * 16) + ti * 16 + lmod];

    for (int t = 0; t < T_STEPS; ++t) {
        // ---- leak + input current: vmem = DECAY*vmem + x[t] + b -------------
        #pragma unroll
        for (int ti = 0; ti < NTILES; ++ti) {
            const int nbase = wave * (NTILES * 16) + ti * 16;
            const float* xp =
                x + ((size_t)t * BATCH + bbase) * (size_t)HID + nbase + lmod;
            if (t + 1 < T_STEPS)
                __builtin_prefetch(xp + (size_t)BATCH * HID, 0, 0); // next step
            #pragma unroll
            for (int r = 0; r < 8; ++r) {
                const int m = r + lhalf * 8;
                acc[ti][r] = DECAY * acc[ti][r] + xp[(size_t)m * HID] + bb[ti];
            }
        }

        // ---- recurrent matmul: vmem += spikes_prev @ (0.5*W)^T --------------
        // K swept in 32-chunks. A from LDS (contiguous: lane<16 holds
        // K[0..15], lane>=16 holds K[16..31] of row M=lmod).
        // B tiles double-buffered in registers, loaded one kt ahead; rolled
        // two-stage loop keeps VGPR pressure bounded (no spills).
        v16bf b0[NTILES], b1[NTILES];
        #pragma unroll
        for (int ti = 0; ti < NTILES; ++ti)       // preload kt = 0
            b0[ti] = *(const v16bf*)(wbase + (size_t)ti * 16 * HID);

        #pragma unroll 1
        for (int kt = 0; kt < HID / 32; kt += 2) {
            union { v16bf v; v8bf h[2]; } a0, a1;
            const int k0 = kt * 32 + lhalf * 16;
            a0.h[0] = *(const v8bf*)&spikes[lmod * HID + k0];
            a0.h[1] = *(const v8bf*)&spikes[lmod * HID + k0 + 8];
            a1.h[0] = *(const v8bf*)&spikes[lmod * HID + k0 + 32];
            a1.h[1] = *(const v8bf*)&spikes[lmod * HID + k0 + 40];

            // issue B loads for kt+1 before consuming b0 (stage 1 -> stage 2)
            #pragma unroll
            for (int ti = 0; ti < NTILES; ++ti)
                b1[ti] = *(const v16bf*)(wbase + (size_t)ti * 16 * HID
                                         + (kt + 1) * 32);
            #pragma unroll
            for (int ti = 0; ti < NTILES; ++ti)
                acc[ti] = __builtin_amdgcn_wmma_f32_16x16x32_bf16(
                    false, a0.v, false, b0[ti], (short)0, acc[ti], false, false);

            // issue B loads for kt+2 before consuming b1 (stage 2 -> stage 1)
            if (kt + 2 < HID / 32) {
                #pragma unroll
                for (int ti = 0; ti < NTILES; ++ti)
                    b0[ti] = *(const v16bf*)(wbase + (size_t)ti * 16 * HID
                                             + (kt + 2) * 32);
            }
            #pragma unroll
            for (int ti = 0; ti < NTILES; ++ti)
                acc[ti] = __builtin_amdgcn_wmma_f32_16x16x32_bf16(
                    false, a1.v, false, b1[ti], (short)0, acc[ti], false, false);
        }

        __syncthreads(); // all waves finished reading spikes_prev from LDS

        // ---- fire, emit output, hard reset, publish new spikes --------------
        #pragma unroll
        for (int ti = 0; ti < NTILES; ++ti) {
            const int nbase = wave * (NTILES * 16) + ti * 16;
            float* op =
                out + ((size_t)t * BATCH + bbase) * (size_t)HID + nbase + lmod;
            #pragma unroll
            for (int r = 0; r < 8; ++r) {
                const int m = r + lhalf * 8;
                const float v = acc[ti][r];
                const bool s = v > THRESH;          // spike_fn(vmem - THRESH)
                op[(size_t)m * HID] = s ? THRESH : 0.0f;
                spikes[m * HID + nbase + lmod] = s ? (bf16_t)1.0f : (bf16_t)0.0f;
                acc[ti][r] = s ? 0.0f : v;          // hard reset
            }
        }
        __syncthreads(); // new spikes visible before next step's A loads
    }
}

// ---- host entry --------------------------------------------------------------
extern "C" void kernel_launch(void* const* d_in, const int* in_sizes, int n_in,
                              void* d_out, int out_size, void* d_ws, size_t ws_size,
                              hipStream_t stream) {
    const float* x  = (const float*)d_in[0]; // [16,1024,1024]
    const float* W  = (const float*)d_in[1]; // [1024,1024]
    const float* b  = (const float*)d_in[2]; // [1024]
    float* out      = (float*)d_out;         // [16,1024,1024]
    bf16_t* Wt      = (bf16_t*)d_ws;         // 2 MB scratch for packed W

    const int nW = HID * HID;
    rlif_pack_w<<<(nW + 255) / 256, 256, 0, stream>>>(W, Wt, nW);
    rlif_step_kernel<<<BATCH / 16, THREADS, 0, stream>>>(x, Wt, b, out);
}
